// SparseGapPerSpecies_75642964017196
// MI455X (gfx1250) — matrix operand order repaired and checked by reading, not compiled, not run on hardware
//
#include <hip/hip_runtime.h>

// ---------------------------------------------------------------------------
// SparseGapPerSpecies on MI455X (gfx1250, wave32, WMMA bf16 16x16x32,
// async global->LDS double-buffered B staging, dual WMMA accumulator chains).
//
// zeta = 2 trick:  sum_m w_m (x_hat . y_hat_m)^2 = (1/||x||^2) sum_m w_m (x . y_hat_m)^2
// so A (power spectrum rows) is used UNNORMALIZED in bf16; the per-row scale
// 1/||x||^2 is applied once at the end. Support rows are pre-normalized.
// ---------------------------------------------------------------------------

#define N_ATOMS    100000
#define N_FEAT     512
#define N_SPECIES  4
#define N_SUPPORT  1000
#define N_STRUCT   2000
#define M_TILES    63                      // ceil(1000/16)
#define SUP_PAD    (M_TILES*16)            // 1008
#define K_STEPS    (N_FEAT/32)             // 16 WMMA k-steps of depth 32
#define N_TILES_TOTAL (N_SPECIES*M_TILES)  // 252
#define TILE_BYTES (K_STEPS*32*32)         // 16384 B per packed B tile (global)
#define LDS_KT_STRIDE 1056                 // 32 lanes*32B + skew pads, per kt
#define LDS_TILE_BYTES (K_STEPS*LDS_KT_STRIDE) // 16896

typedef __attribute__((ext_vector_type(16))) __bf16 v16bf;
typedef __attribute__((ext_vector_type(8)))  __bf16 v8bf;
typedef __attribute__((ext_vector_type(8)))  float  v8f;

__device__ __forceinline__ unsigned short f32_to_bf16(float f) {
  unsigned u = __float_as_uint(f);
  u += 0x7FFFu + ((u >> 16) & 1u);   // round-to-nearest-even
  return (unsigned short)(u >> 16);
}

// CDNA5 async global->LDS copy (16B per lane), tracked by ASYNCcnt.
__device__ __forceinline__ void async_load_b128(const void* gaddr, unsigned lds_off) {
  asm volatile("global_load_async_to_lds_b128 %0, %1, off"
               :: "v"(lds_off), "v"(gaddr) : "memory");
}
__device__ __forceinline__ void wait_asynccnt0() {
  asm volatile("s_wait_asynccnt 0x0" ::: "memory");
}

// ---------------------------------------------------------------------------
// Kernel 1: normalize support rows, pad 1000->1008, pre-pack into per-lane
// WMMA-B fragment order: for (kt, lane), 16 bf16 contiguous (32B).
// B k order used here (contiguous, same as A; not AMD's table order): column N = lane%16; lanes 0-15 hold
// K 0..15 of the slice, lanes 16-31 hold K 16..31.
// ---------------------------------------------------------------------------
__global__ void prep_support_kernel(const float* __restrict__ sp,
                                    const float* __restrict__ w,
                                    unsigned short* __restrict__ Bpack,
                                    float* __restrict__ wpad) {
  int wave = (int)((blockIdx.x * blockDim.x + threadIdx.x) >> 5);
  int lane = threadIdx.x & 31;
  if (wave >= N_SPECIES * SUP_PAD) return;
  int s  = wave / SUP_PAD;
  int m  = wave - s * SUP_PAD;
  int mt = m >> 4, col = m & 15;

  float vals[16];
  float sum = 0.f;
  if (m < N_SUPPORT) {
    const float* row = sp + (size_t)(s * N_SUPPORT + m) * N_FEAT + lane * 16;
    #pragma unroll
    for (int i = 0; i < 16; ++i) { vals[i] = row[i]; sum += vals[i] * vals[i]; }
  } else {
    #pragma unroll
    for (int i = 0; i < 16; ++i) vals[i] = 0.f;
  }
  #pragma unroll
  for (int off = 16; off >= 1; off >>= 1) sum += __shfl_xor(sum, off, 32);
  float inv = (sum > 0.f) ? rsqrtf(sum) : 0.f;

  int kt      = lane >> 1;                       // lane covers K = lane*16..+15
  int dstlane = col + ((lane & 1) << 4);
  size_t off  = (((size_t)(s * M_TILES + mt) * K_STEPS + kt) * 32 + dstlane) * 16;

  unsigned short tmp[16] __attribute__((aligned(16)));
  #pragma unroll
  for (int i = 0; i < 16; ++i) tmp[i] = f32_to_bf16(vals[i] * inv);
  *(uint4*)(Bpack + off)     = *(const uint4*)(tmp);
  *(uint4*)(Bpack + off + 8) = *(const uint4*)(tmp + 8);

  if (lane == 0)
    wpad[s * SUP_PAD + m] = (m < N_SUPPORT) ? w[s * N_SUPPORT + m] : 0.f;
}

// ---------------------------------------------------------------------------
// Kernel 2: WMMA GEMM. 4 waves/block, one 16-atom tile per wave; all four
// waves share each 16KB B tile, staged into LDS with double-buffered async
// global->LDS copies (8 x b128 per thread per tile). K is split across two
// independent WMMA accumulator chains (c0/c1) so consecutive WMMAs have no
// RAW dependency and can issue back-to-back even at low occupancy.
// A fragment layout (16-bit A 16x32): row M = lane&15; lane<16 holds
// K 0-15 of the slice, lane>=16 holds K 16-31 (this code's contiguous order, as in Bpack).
// ---------------------------------------------------------------------------
__global__ void __launch_bounds__(128)
gap_gemm_kernel(const float* __restrict__ X,
                const unsigned short* __restrict__ Bpack,
                const float* __restrict__ wpad,
                const int* __restrict__ species,
                float* __restrict__ e_atom) {
  __shared__ __align__(16) unsigned char bbuf[2][LDS_TILE_BYTES];

  int tid  = threadIdx.x;
  int wave = tid >> 5, lane = tid & 31;
  int tile_base = ((int)blockIdx.x * 4 + wave) * 16;
  int row_local = lane & 15;
  int half      = lane >> 4;
  int row       = tile_base + row_local;
  bool valid    = row < N_ATOMS;

  // Per-thread async-copy slots: thread -> (kt = tid/8), 8 16B chunks each.
  // LDS layout skews each lane's 32B by (lane/8)*16B -> 2-way bank access.
  int kt_t = tid >> 3, sub = tid & 7;
  unsigned goff[8], loff[8];
  #pragma unroll
  for (int j = 0; j < 8; ++j) {
    int q = sub * 8 + j, lane_l = q >> 1, hf = q & 1;
    goff[j] = (unsigned)(((kt_t * 32 + lane_l) * 16 + hf * 8) * 2);         // bytes in global tile
    loff[j] = (unsigned)(kt_t * LDS_KT_STRIDE + lane_l * 32 + (lane_l >> 3) * 16 + hf * 16);
  }
  unsigned lds_b0 = (unsigned)(unsigned long long)(&bbuf[0][0]);
  unsigned lds_b1 = (unsigned)(unsigned long long)(&bbuf[1][0]);

  // Prefetch tile 0 into buffer 0 (overlaps the A-fragment global loads).
  {
    const char* g = (const char*)Bpack;
    #pragma unroll
    for (int j = 0; j < 8; ++j) async_load_b128(g + goff[j], lds_b0 + loff[j]);
  }

  // ---- A fragments straight from global (unnormalized bf16) + row sum-sq.
  const float* xrow = X + (size_t)(valid ? row : 0) * N_FEAT;
  v16bf a_frag[K_STEPS];
  float sumsq = 0.f;
  #pragma unroll
  for (int kt = 0; kt < K_STEPS; ++kt) {
    int c0 = kt * 32 + (half ? 16 : 0);
    float4 f0 = *(const float4*)(xrow + c0);
    float4 f1 = *(const float4*)(xrow + c0 + 4);
    float4 f2 = *(const float4*)(xrow + c0 + 8);
    float4 f3 = *(const float4*)(xrow + c0 + 12);
    sumsq += f0.x*f0.x + f0.y*f0.y + f0.z*f0.z + f0.w*f0.w
           + f1.x*f1.x + f1.y*f1.y + f1.z*f1.z + f1.w*f1.w
           + f2.x*f2.x + f2.y*f2.y + f2.z*f2.z + f2.w*f2.w
           + f3.x*f3.x + f3.y*f3.y + f3.z*f3.z + f3.w*f3.w;
    union { unsigned short u[16]; v16bf v; } af;
    af.u[0]=f32_to_bf16(f0.x);  af.u[1]=f32_to_bf16(f0.y);
    af.u[2]=f32_to_bf16(f0.z);  af.u[3]=f32_to_bf16(f0.w);
    af.u[4]=f32_to_bf16(f1.x);  af.u[5]=f32_to_bf16(f1.y);
    af.u[6]=f32_to_bf16(f1.z);  af.u[7]=f32_to_bf16(f1.w);
    af.u[8]=f32_to_bf16(f2.x);  af.u[9]=f32_to_bf16(f2.y);
    af.u[10]=f32_to_bf16(f2.z); af.u[11]=f32_to_bf16(f2.w);
    af.u[12]=f32_to_bf16(f3.x); af.u[13]=f32_to_bf16(f3.y);
    af.u[14]=f32_to_bf16(f3.z); af.u[15]=f32_to_bf16(f3.w);
    a_frag[kt] = af.v;
  }
  float tot  = sumsq + __shfl_xor(sumsq, 16, 32);
  float inv2 = (tot > 0.f) ? (1.0f / tot) : 0.f;   // 1/||x||^2 for my row

  float acc[N_SPECIES][8];
  #pragma unroll
  for (int s = 0; s < N_SPECIES; ++s)
    #pragma unroll
    for (int v = 0; v < 8; ++v) acc[s][v] = 0.f;

  unsigned myoff = (unsigned)(lane * 32 + (lane >> 3) * 16);

  #pragma unroll
  for (int s = 0; s < N_SPECIES; ++s) {
    for (int mt = 0; mt < M_TILES; ++mt) {
      int tl = s * M_TILES + mt;

      wait_asynccnt0();      // tile tl landed in LDS (this thread's portion)
      __syncthreads();       // everyone's portion landed; prev-buffer readers done

      if (tl + 1 < N_TILES_TOTAL) {   // prefetch next tile into other buffer
        const char* g = (const char*)Bpack + (size_t)(tl + 1) * TILE_BYTES;
        unsigned lb = ((tl + 1) & 1) ? lds_b1 : lds_b0;
        #pragma unroll
        for (int j = 0; j < 8; ++j) async_load_b128(g + goff[j], lb + loff[j]);
      }

      const unsigned char* buf = bbuf[tl & 1];
      float wl = wpad[s * SUP_PAD + mt * 16 + row_local];  // weight of my column

      // Two independent accumulator chains over even/odd k-steps.
      v8f c0 = {}, c1 = {};
      #pragma unroll
      for (int kt = 0; kt < K_STEPS; kt += 2) {
        {
          const v8bf* p = (const v8bf*)(buf + kt * LDS_KT_STRIDE + myoff);
          v8bf lo = p[0], hi = p[1];
          v16bf b;
          #pragma unroll
          for (int i = 0; i < 8; ++i) { b[i] = lo[i]; b[i + 8] = hi[i]; }
          c0 = __builtin_amdgcn_wmma_f32_16x16x32_bf16(
                  false, a_frag[kt], false, b, (short)0, c0, false, false);
        }
        {
          const v8bf* p = (const v8bf*)(buf + (kt + 1) * LDS_KT_STRIDE + myoff);
          v8bf lo = p[0], hi = p[1];
          v16bf b;
          #pragma unroll
          for (int i = 0; i < 8; ++i) { b[i] = lo[i]; b[i + 8] = hi[i]; }
          c1 = __builtin_amdgcn_wmma_f32_16x16x32_bf16(
                  false, a_frag[kt + 1], false, b, (short)0, c1, false, false);
        }
      }
      // zeta = 2: e += w[col] * K^2, K = full-depth dot (c0+c1).
      // (C layout: lane = col, vgpr v = row)
      #pragma unroll
      for (int v = 0; v < 8; ++v) {
        float k = c0[v] + c1[v];
        acc[s][v] += wl * k * k;
      }
    }
  }

  // Per-row species select, 1/||x||^2 scale, column reduction, store.
  #pragma unroll
  for (int v = 0; v < 8; ++v) {
    int m_row = v + (half ? 8 : 0);
    int atom  = tile_base + m_row;
    int sp    = (atom < N_ATOMS) ? species[atom] : 0;
    float e = (sp == 0) ? acc[0][v]
            : (sp == 1) ? acc[1][v]
            : (sp == 2) ? acc[2][v]
                        : acc[3][v];
    e *= __shfl(inv2, v + (half << 3), 32);        // scale of row m_row
    #pragma unroll
    for (int off = 8; off >= 1; off >>= 1) e += __shfl_xor(e, off, 32);
    if ((lane & 15) == 0 && atom < N_ATOMS) e_atom[atom] = e;
  }
}

// ---------------------------------------------------------------------------
// Kernel 3: deterministic segment sum (structure_ids sorted).
// One wave per structure: binary-search range, fixed-tree reduce.
// ---------------------------------------------------------------------------
__global__ void segment_sum_kernel(const float* __restrict__ e_atom,
                                   const int* __restrict__ sids,
                                   float* __restrict__ out) {
  int wave = (int)((blockIdx.x * blockDim.x + threadIdx.x) >> 5);
  int lane = threadIdx.x & 31;
  if (wave >= N_STRUCT) return;

  int lo = 0, hi = N_ATOMS;
  while (lo < hi) { int mid = (lo + hi) >> 1; if (sids[mid] < wave) lo = mid + 1; else hi = mid; }
  int start = lo;
  hi = N_ATOMS;
  while (lo < hi) { int mid = (lo + hi) >> 1; if (sids[mid] < wave + 1) lo = mid + 1; else hi = mid; }
  int end = lo;

  float sum = 0.f;
  for (int i = start + lane; i < end; i += 32) sum += e_atom[i];
  #pragma unroll
  for (int off = 16; off >= 1; off >>= 1) sum += __shfl_xor(sum, off, 32);
  if (lane == 0) out[wave] = sum;
}

// ---------------------------------------------------------------------------
extern "C" void kernel_launch(void* const* d_in, const int* in_sizes, int n_in,
                              void* d_out, int out_size, void* d_ws, size_t ws_size,
                              hipStream_t stream) {
  const float* X       = (const float*)d_in[0];  // [100000,512] f32
  const float* SP      = (const float*)d_in[1];  // [4,1000,512] f32
  const float* W       = (const float*)d_in[2];  // [4,1000] f32
  const int*   species = (const int*)d_in[3];    // [100000] i32
  const int*   sids    = (const int*)d_in[4];    // [100000] i32 (sorted)
  (void)in_sizes; (void)n_in; (void)out_size; (void)ws_size;

  // Workspace layout
  size_t bpack_elems = (size_t)N_TILES_TOTAL * K_STEPS * 32 * 16;  // bf16 elems
  unsigned short* Bpack = (unsigned short*)d_ws;
  float* wpad   = (float*)((char*)d_ws + bpack_elems * sizeof(unsigned short));
  float* e_atom = (float*)((char*)wpad + (size_t)N_SPECIES * SUP_PAD * sizeof(float));

  // 1) pack + normalize support set (4032 waves)
  int prep_threads = N_SPECIES * SUP_PAD * 32;
  prep_support_kernel<<<(prep_threads + 255) / 256, 256, 0, stream>>>(SP, W, Bpack, wpad);

  // 2) WMMA GEMM: 6250 atom tiles, 4 tiles (waves) per block sharing B via LDS
  int n_tiles  = (N_ATOMS + 15) / 16;
  int n_blocks = (n_tiles + 3) / 4;
  gap_gemm_kernel<<<n_blocks, 128, 0, stream>>>(X, Bpack, wpad, species, e_atom);

  // 3) deterministic per-structure sum
  segment_sum_kernel<<<(N_STRUCT * 32 + 255) / 256, 256, 0, stream>>>(e_atom, sids, (float*)d_out);
}
